// MSATransformerBlock_63608465654537
// MI455X (gfx1250) — matrix operand order repaired and checked by reading, not compiled, not run on hardware
//
#include <hip/hip_runtime.h>

typedef unsigned short u16;
typedef unsigned int   u32;
typedef unsigned long long u64;

typedef __bf16 bf16x16 __attribute__((ext_vector_type(16)));
typedef float  f32x8   __attribute__((ext_vector_type(8)));
typedef u32    uint4v  __attribute__((ext_vector_type(4)));
typedef int    int4v   __attribute__((ext_vector_type(4)));
typedef int    int8v   __attribute__((ext_vector_type(8)));

union Frag {
  bf16x16 v;
  uint4v  q[2];
  u32     u[8];
};

__device__ __forceinline__ u16 f2bf(float f) {
  u32 x = __float_as_uint(f);
  x += 0x7FFFu + ((x >> 16) & 1u);          // round-to-nearest-even
  return (u16)(x >> 16);
}

__device__ __forceinline__ f32x8 wmma_bf16(const Frag& a, const Frag& b, f32x8 c) {
  return __builtin_amdgcn_wmma_f32_16x16x32_bf16(false, a.v, false, b.v,
                                                 (short)0, c, false, false);
}

__device__ __forceinline__ int map_row(int r, int n1, int n2) {
  return n1 ? (r % n1) * n2 + (r / n1) : r;
}

// ---------------------------------------------------------------------------
// Tensor Data Mover: 2-D tile (bf16) global -> LDS.  D# per ISA 08 §8.
// ---------------------------------------------------------------------------
#if defined(__gfx1250__) && __has_builtin(__builtin_amdgcn_tensor_load_to_lds)
#define HAVE_TDM 1
#else
#define HAVE_TDM 0
#endif

#if HAVE_TDM
__device__ __forceinline__ void tdm_load_2d(u32 lds_off, const void* gaddr,
                                            u32 tensor_d0, u32 tensor_d1,
                                            u32 tile_d0, u32 tile_d1, u32 stride_d0) {
  u64 ga = (u64)(uintptr_t)gaddr;
  uint4v g0;
  g0.x = 1u;                                            // count=1 (user descriptor)
  g0.y = lds_off;                                       // lds_addr [63:32]
  g0.z = (u32)ga;                                       // global_addr [95:64]
  g0.w = (u32)((ga >> 32) & 0x1FFFFFFu) | (2u << 30);   // addr[56:32] | type=2
  int8v g1;
  g1[0] = (int)0x00010000u;                             // data_size=1 (2 bytes)
  g1[1] = (int)((tensor_d0 & 0xFFFFu) << 16);           // tensor_dim0 lo16 @48
  g1[2] = (int)(((tensor_d0 >> 16) & 0xFFFFu) | ((tensor_d1 & 0xFFFFu) << 16));
  g1[3] = (int)(((tensor_d1 >> 16) & 0xFFFFu) | ((tile_d0 & 0xFFFFu) << 16));
  g1[4] = (int)(tile_d1 & 0xFFFFu);                     // tile_dim1, tile_dim2=0
  g1[5] = (int)stride_d0;                               // tensor_dim0_stride[31:0]
  g1[6] = 0;                                            // stride hi, dim1_stride lo
  g1[7] = 0;
  int4v z4 = {0, 0, 0, 0};
#if __has_include(<hip/amd_detail/amd_gfx1250_TDM.h>)
  int8v z8 = {0, 0, 0, 0, 0, 0, 0, 0};
  __builtin_amdgcn_tensor_load_to_lds(g0, g1, z4, z4, z8, 0);
#else
  __builtin_amdgcn_tensor_load_to_lds(g0, g1, z4, z4, 0);
#endif
}
#endif

// ---------------------------------------------------------------------------
// fp32 [K][N] -> bf16 transposed [N][K] (weights become B-fragment friendly)
// ---------------------------------------------------------------------------
__global__ void cvt_t_kernel(const float* __restrict__ in, u16* __restrict__ out,
                             int Kd, int Nd) {
  int i = blockIdx.x * 256 + threadIdx.x;
  if (i < Kd * Nd) {
    int k = i / Nd, nn = i % Nd;
    out[(size_t)nn * Kd + k] = f2bf(in[i]);
  }
}

// ---------------------------------------------------------------------------
// LayerNorm: fp32 in -> bf16 out, optional output-row transpose remap
// ---------------------------------------------------------------------------
__global__ __launch_bounds__(128) void ln_bf16_kernel(
    const float* __restrict__ x, const float* __restrict__ g, const float* __restrict__ b,
    u16* __restrict__ out, int D, int n1, int n2)
{
  __shared__ float red[128];
  const int r = blockIdx.x;
  const int t = threadIdx.x;
  const float* xr = x + (size_t)r * D;

  float s = 0.f;
  for (int i = t; i < D; i += 128) s += xr[i];
  red[t] = s; __syncthreads();
  for (int o = 64; o > 0; o >>= 1) { if (t < o) red[t] += red[t + o]; __syncthreads(); }
  float mean = red[0] / (float)D;
  __syncthreads();

  float s2 = 0.f;
  for (int i = t; i < D; i += 128) { float d = xr[i] - mean; s2 += d * d; }
  red[t] = s2; __syncthreads();
  for (int o = 64; o > 0; o >>= 1) { if (t < o) red[t] += red[t + o]; __syncthreads(); }
  float rstd = rsqrtf(red[0] / (float)D + 1e-5f);

  u16* orow = out + (size_t)map_row(r, n1, n2) * D;
  for (int i = t; i < D; i += 128)
    orow[i] = f2bf((xr[i] - mean) * rstd * g[i] + b[i]);
}

// ---------------------------------------------------------------------------
// GEMM: C[M,N] = act(A[M,K](bf16) * W[K,N] + bias) (+resid), W given as Wt[N][K].
// 256 thr = 8 waves; block tile 32(M) x 128(N); wave tile 16x32 (2 acc).
// Tiles staged by TDM (double buffered, s_wait_tensorcnt) when available.
// All fragments are 2x ds_load_b128 per matrix.
// ---------------------------------------------------------------------------
#define GEMM_FRAG_STEP(BUF)                                                   \
  do {                                                                        \
    Frag a_, b0_, b1_;                                                        \
    a_.q[0]  = *(const uint4v*)&As[BUF][mt + l16][hf * 16];                   \
    a_.q[1]  = *(const uint4v*)&As[BUF][mt + l16][hf * 16 + 8];               \
    b0_.q[0] = *(const uint4v*)&Bs[BUF][ntb + l16][hf * 16];                  \
    b0_.q[1] = *(const uint4v*)&Bs[BUF][ntb + l16][hf * 16 + 8];              \
    b1_.q[0] = *(const uint4v*)&Bs[BUF][ntb + 16 + l16][hf * 16];             \
    b1_.q[1] = *(const uint4v*)&Bs[BUF][ntb + 16 + l16][hf * 16 + 8];         \
    acc0 = wmma_bf16(a_, b0_, acc0);                                          \
    acc1 = wmma_bf16(a_, b1_, acc1);                                          \
  } while (0)

__global__ __launch_bounds__(256) void gemm_bf16_kernel(
    const u16* __restrict__ A, int lda,
    const u16* __restrict__ Wt,
    const float* __restrict__ bias,
    const float* __restrict__ resid,
    float* __restrict__ outf,
    u16* __restrict__ outb,
    int M, int N, int K, int ldo, int n1, int n2, int relu)
{
  __shared__ __align__(16) u16 As[2][32][32];
  __shared__ __align__(16) u16 Bs[2][128][32];

  const int tid  = threadIdx.x;
  const int lane = tid & 31;
  const int wave = tid >> 5;
  const int hf   = lane >> 4;
  const int l16  = lane & 15;
  const int m0   = blockIdx.y * 32;
  const int n0   = blockIdx.x * 128;
  const int mt   = (wave & 1) * 16;
  const int ntb  = (wave >> 1) * 32;
  const int nsteps = K >> 5;

  f32x8 acc0 = {}, acc1 = {};

#if HAVE_TDM
  const u32 lds_a = (u32)(uintptr_t)&As[0][0][0];
  const u32 lds_b = (u32)(uintptr_t)&Bs[0][0][0];
  const u32 bd1 = (u32)((N - n0 < 128) ? (N - n0) : 128);
  if (tid < 32) {
    tdm_load_2d(lds_a, A + (size_t)m0 * lda, (u32)K, (u32)(M - m0), 32, 32, (u32)lda);
    tdm_load_2d(lds_b, Wt + (size_t)n0 * K, (u32)K, bd1, 32, bd1, (u32)K);
  }
  for (int s = 0; s < nsteps; ++s) {
    const int buf = s & 1;
    if (s + 1 < nsteps) {
      const int kn = (s + 1) << 5;
      if (tid < 32) {
        tdm_load_2d(lds_a + (u32)(buf ^ 1) * (u32)sizeof(As[0]),
                    A + (size_t)m0 * lda + kn, (u32)(K - kn), (u32)(M - m0), 32, 32, (u32)lda);
        tdm_load_2d(lds_b + (u32)(buf ^ 1) * (u32)sizeof(Bs[0]),
                    Wt + (size_t)n0 * K + kn, (u32)(K - kn), bd1, 32, bd1, (u32)K);
        __builtin_amdgcn_s_wait_tensorcnt(2);   // current buffer resident
      }
    } else {
      if (tid < 32) __builtin_amdgcn_s_wait_tensorcnt(0);
    }
    __syncthreads();
    GEMM_FRAG_STEP(buf);
    __syncthreads();
  }
#else
  for (int s = 0; s < nsteps; ++s) {
    const int kk = s << 5;
    for (int idx = tid; idx < 32 * 32; idx += 256) {
      int r = idx >> 5, c = idx & 31;
      int gr = m0 + r;
      As[0][r][c] = (gr < M) ? A[(size_t)gr * lda + kk + c] : (u16)0;
    }
    for (int idx = tid; idx < 128 * 32; idx += 256) {
      int r = idx >> 5, c = idx & 31;
      int gn = n0 + r;
      Bs[0][r][c] = (gn < N) ? Wt[(size_t)gn * K + kk + c] : (u16)0;
    }
    __syncthreads();
    GEMM_FRAG_STEP(0);
    __syncthreads();
  }
#endif

  // epilogue
#pragma unroll
  for (int p = 0; p < 2; ++p) {
    const f32x8 acc = p ? acc1 : acc0;
    const int col = n0 + ntb + p * 16 + l16;
    if (col < N) {
#pragma unroll
      for (int r = 0; r < 8; ++r) {
        int gm = m0 + mt + hf * 8 + r;
        if (gm < M) {
          int orow = map_row(gm, n1, n2);
          float v = acc[r];
          if (bias)  v += bias[col];
          if (relu)  v = fmaxf(v, 0.f);
          if (resid) v += resid[(size_t)orow * ldo + col];
          if (outf)  outf[(size_t)orow * ldo + col] = v;
          if (outb)  outb[(size_t)orow * ldo + col] = f2bf(v);
        }
      }
    }
  }
}

// ---------------------------------------------------------------------------
// Fused attention: scores = clip(QK^T * scale), softmax, ctx = P V
// qkv: bf16 [R*n, 3*dm], head_dim 32.  One wave per (r, head, 16-row block)
// ---------------------------------------------------------------------------
__global__ __launch_bounds__(32) void attn_bf16_kernel(
    const u16* __restrict__ qkv, u16* __restrict__ ctx,
    int n, int heads, int dm, float scale)
{
  __shared__ __align__(16) float sc[16][256];
  __shared__ __align__(16) u16   pb[16][256];

  const int ntiles = n >> 4;
  const int wg   = blockIdx.x;
  const int it   = wg % ntiles;
  const int h    = (wg / ntiles) % heads;
  const int r    = wg / (ntiles * heads);
  const int lane = threadIdx.x;
  const int hf   = lane >> 4;
  const int l16  = lane & 15;
  const int i0   = it * 16;
  const int row3 = 3 * dm;
  const int kb1  = hf * 8, kb2 = 16 + hf * 8, ko = hf * 16;

  Frag aq;
  {
    const u16* qrow = qkv + (size_t)(r * n + i0 + l16) * row3 + h * 32;
    aq.q[0] = *(const uint4v*)(qrow + ko);
    aq.q[1] = *(const uint4v*)(qrow + ko + 8);
  }

#pragma unroll 1
  for (int jt = 0; jt < ntiles; ++jt) {
    Frag bk;
    const u16* krow = qkv + (size_t)(r * n + jt * 16 + l16) * row3 + dm + h * 32;
    bk.q[0] = *(const uint4v*)(krow + ko);
    bk.q[1] = *(const uint4v*)(krow + ko + 8);
    f32x8 c = {};
    c = wmma_bf16(aq, bk, c);
#pragma unroll
    for (int rr = 0; rr < 8; ++rr) {
      float s = c[rr] * scale;
      sc[hf * 8 + rr][jt * 16 + l16] = fminf(fmaxf(s, -50.f), 50.f);
    }
  }
  __syncthreads();

  if (lane < 16) {
    float m = -1e30f;
    for (int j = 0; j < n; ++j) m = fmaxf(m, sc[lane][j]);
    float s = 0.f;
    for (int j = 0; j < n; ++j) { float e = __expf(sc[lane][j] - m); sc[lane][j] = e; s += e; }
    float inv = 1.f / s;
    for (int j = 0; j < n; ++j) pb[lane][j] = f2bf(sc[lane][j] * inv);
  }
  __syncthreads();

#pragma unroll 1
  for (int t = 0; t < 2; ++t) {
    f32x8 c = {};
#pragma unroll 1
    for (int kk = 0; kk < n; kk += 32) {
      Frag ap, bv;
      ap.q[0] = *(const uint4v*)&pb[l16][kk + ko];
      ap.q[1] = *(const uint4v*)&pb[l16][kk + ko + 8];
      const u16* vb = qkv + (size_t)(r * n + kk) * row3 + 2 * dm + h * 32 + t * 16 + l16;
#pragma unroll
      for (int v = 0; v < 8; ++v) {
        u32 lo = vb[(size_t)(ko + 2 * v) * row3];
        u32 hi = vb[(size_t)(ko + 2 * v + 1) * row3];
        bv.u[v] = lo | (hi << 16);
      }
      c = wmma_bf16(ap, bv, c);
    }
    u16* crow = ctx + (size_t)(r * n + i0 + hf * 8) * dm + h * 32 + t * 16 + l16;
#pragma unroll
    for (int rr = 0; rr < 8; ++rr) crow[(size_t)rr * dm] = f2bf(c[rr]);
  }
}

// ---------------------------------------------------------------------------
// Fused outer-product-mean + 1024->DP projection + pair residual.
// Wop given transposed: [DP][1024].
// ---------------------------------------------------------------------------
__global__ __launch_bounds__(32) void outer_proj_kernel(
    const u16* __restrict__ abuf, const u16* __restrict__ bbuf,
    const u16* __restrict__ Wop, const float* __restrict__ bias,
    const float* __restrict__ pair_in, float* __restrict__ pair_out,
    int L_, int S_, int NO_, int DP_)
{
  __shared__ __align__(16) u16 O[32][512];

  const int jtiles = L_ / 16;
  const int jt = blockIdx.x % jtiles;
  const int i  = blockIdx.x / jtiles;
  const int j0 = jt * 16;
  const int lane = threadIdx.x;
  const int hf = lane >> 4, l16 = lane & 15;
  const int kb1 = hf * 16, kb2 = hf * 16 + 8;
  const float invS = 1.0f / (float)S_;
  const size_t rowstride = (size_t)L_ * NO_;

  Frag afr[2][2];
#pragma unroll
  for (int mt = 0; mt < 2; ++mt)
#pragma unroll
    for (int kq = 0; kq < 2; ++kq) {
      const int kk = kq * 32;
      const int crow = mt * 16 + l16;
#pragma unroll
      for (int v = 0; v < 4; ++v) {
        int s1 = kk + kb1 + 2 * v, s2 = kk + kb2 + 2 * v;
        afr[mt][kq].u[v] =
            (u32)abuf[(size_t)s1 * rowstride + (size_t)i * NO_ + crow] |
            ((u32)abuf[(size_t)(s1 + 1) * rowstride + (size_t)i * NO_ + crow] << 16);
        afr[mt][kq].u[4 + v] =
            (u32)abuf[(size_t)s2 * rowstride + (size_t)i * NO_ + crow] |
            ((u32)abuf[(size_t)(s2 + 1) * rowstride + (size_t)i * NO_ + crow] << 16);
      }
    }

#pragma unroll 1
  for (int nt = 0; nt < 32; ++nt) {
    const int ncol = nt * 16 + l16;
    const int jl = ncol >> 5, d = ncol & 31;
    const u16* bb = bbuf + (size_t)(j0 + jl) * NO_ + d;
#pragma unroll 1
    for (int mt = 0; mt < 2; ++mt) {
      f32x8 c = {};
#pragma unroll
      for (int kq = 0; kq < 2; ++kq) {
        Frag bfr;
        const int ko = kq * 32 + hf * 16;
#pragma unroll
        for (int v = 0; v < 8; ++v) {
          u32 lo = bb[(size_t)(ko + 2 * v) * rowstride];
          u32 hi = bb[(size_t)(ko + 2 * v + 1) * rowstride];
          bfr.u[v] = lo | (hi << 16);
        }
        c = wmma_bf16(afr[mt][kq], bfr, c);
      }
#pragma unroll
      for (int rr = 0; rr < 8; ++rr)
        O[mt * 16 + hf * 8 + rr][nt * 16 + l16] = f2bf(c[rr] * invS);
    }
  }
  __syncthreads();

#pragma unroll 1
  for (int nt = 0; nt < 8; ++nt) {
    const int pcol = nt * 16 + l16;
    const u16* wr = Wop + (size_t)pcol * 1024;
    f32x8 c = {};
#pragma unroll 1
    for (int kk = 0; kk < 1024; kk += 32) {
      Frag av, bw;
      av.q[0] = *(const uint4v*)&O[kk >> 5][l16 * 32 + kb1];
      av.q[1] = *(const uint4v*)&O[kk >> 5][l16 * 32 + kb2];
      bw.q[0] = *(const uint4v*)(wr + kk + hf * 16);
      bw.q[1] = *(const uint4v*)(wr + kk + hf * 16 + 8);
      c = wmma_bf16(av, bw, c);
    }
#pragma unroll
    for (int rr = 0; rr < 8; ++rr) {
      int j = j0 + hf * 8 + rr;
      size_t idx = ((size_t)i * L_ + j) * DP_ + pcol;
      pair_out[idx] = c[rr] + bias[pcol] + pair_in[idx];
    }
  }
}

// ---------------------------------------------------------------------------
extern "C" void kernel_launch(void* const* d_in, const int* in_sizes, int n_in,
                              void* d_out, int out_size, void* d_ws, size_t ws_size,
                              hipStream_t stream) {
  (void)in_sizes; (void)n_in; (void)out_size; (void)ws_size;
  const int S = 64, L = 256, DM = 256, DP = 128, H = 8, HT = 4, NO = 32;
  const float scale32 = 0.1767766953f;  // 1/sqrt(32)
  #define FP(i) ((const float*)d_in[i])

  const float* msa_in  = FP(0);
  const float* pair_in = FP(1);

  char* ws = (char*)d_ws;
  size_t off = 0;
  auto alloc = [&](size_t bytes) -> void* {
    void* p = ws + off;
    off = (off + bytes + 255) & ~(size_t)255;
    return p;
  };

  float* msa_cur  = (float*)alloc((size_t)S * L * DM * 4);
  float* pair_cur = (float*)alloc((size_t)L * L * DP * 4);
  u16*   xln      = (u16*)alloc((size_t)L * L * DP * 2);
  u16*   qkv      = (u16*)alloc((size_t)L * L * 3 * DP * 2);
  u16*   ctx      = (u16*)alloc((size_t)L * L * DP * 2);
  u16*   h1       = (u16*)alloc((size_t)S * L * 4 * DM * 2);
  u16*   a_bf     = (u16*)alloc((size_t)S * L * NO * 2);
  u16*   b_bf     = (u16*)alloc((size_t)S * L * NO * 2);

  // weights -> bf16, transposed to [N][K]
  auto cvt = [&](int idx, int Kd, int Nd) -> u16* {
    u16* dst = (u16*)alloc((size_t)Kd * Nd * 2);
    cvt_t_kernel<<<(Kd * Nd + 255) / 256, 256, 0, stream>>>(FP(idx), dst, Kd, Nd);
    return dst;
  };
  u16* w_row_qkv = cvt(4,  DM, 3 * DM);
  u16* w_row_out = cvt(6,  DM, DM);
  u16* w_col_qkv = cvt(10, DM, 3 * DM);
  u16* w_col_out = cvt(12, DM, DM);
  u16* w_ff1     = cvt(16, DM, 4 * DM);
  u16* w_ff2     = cvt(18, 4 * DM, DM);
  u16* w_opa     = cvt(22, DM, NO);
  u16* w_opb     = cvt(24, DM, NO);
  u16* w_opout   = cvt(26, NO * NO, DP);
  u16* w_tri_qkv = cvt(30, DP, 3 * DP);
  u16* w_tri_out = cvt(32, DP, DP);

  auto ln = [&](const float* x, int gi, int bi, u16* out, int R, int D, int n1, int n2) {
    ln_bf16_kernel<<<R, 128, 0, stream>>>(x, FP(gi), FP(bi), out, D, n1, n2);
  };
  auto gemm = [&](const u16* A, int lda, const u16* Wt, const float* bias,
                  const float* resid, float* outf, u16* outb,
                  int M, int N, int K, int ldo, int n1, int n2, int relu) {
    dim3 g((N + 127) / 128, M / 32);
    gemm_bf16_kernel<<<g, 256, 0, stream>>>(A, lda, Wt, bias, resid, outf, outb,
                                            M, N, K, ldo, n1, n2, relu);
  };
  auto attn = [&](const u16* q, u16* c, int R, int n, int heads, int dm) {
    attn_bf16_kernel<<<R * heads * (n / 16), 32, 0, stream>>>(q, c, n, heads, dm, scale32);
  };

  // ---- row attention (over L) ----
  ln(msa_in, 2, 3, xln, S * L, DM, 0, 0);
  gemm(xln, DM, w_row_qkv, FP(5), nullptr, nullptr, qkv, S * L, 3 * DM, DM, 3 * DM, 0, 0, 0);
  attn(qkv, ctx, S, L, H, DM);
  gemm(ctx, DM, w_row_out, FP(7), msa_in, msa_cur, nullptr, S * L, DM, DM, DM, 0, 0, 0);

  // ---- column attention (over S), transpose folded into remaps ----
  ln(msa_cur, 8, 9, xln, S * L, DM, L, S);
  gemm(xln, DM, w_col_qkv, FP(11), nullptr, nullptr, qkv, L * S, 3 * DM, DM, 3 * DM, 0, 0, 0);
  attn(qkv, ctx, L, S, H, DM);
  gemm(ctx, DM, w_col_out, FP(13), msa_cur, msa_cur, nullptr, L * S, DM, DM, DM, S, L, 0);

  // ---- feed-forward ----
  ln(msa_cur, 14, 15, xln, S * L, DM, 0, 0);
  gemm(xln, DM, w_ff1, FP(17), nullptr, nullptr, h1, S * L, 4 * DM, DM, 4 * DM, 0, 0, 1);
  gemm(h1, 4 * DM, w_ff2, FP(19), msa_cur, msa_cur, nullptr, S * L, DM, 4 * DM, DM, 0, 0, 0);

  // ---- outer product mean -> pair ----
  ln(msa_cur, 20, 21, xln, S * L, DM, 0, 0);
  gemm(xln, DM, w_opa, FP(23), nullptr, nullptr, a_bf, S * L, NO, DM, NO, 0, 0, 0);
  gemm(xln, DM, w_opb, FP(25), nullptr, nullptr, b_bf, S * L, NO, DM, NO, 0, 0, 0);
  outer_proj_kernel<<<L * (L / 16), 32, 0, stream>>>(a_bf, b_bf, w_opout, FP(27),
                                                     pair_in, pair_cur, L, S, NO, DP);

  // ---- triangle attention on pair ----
  ln(pair_cur, 28, 29, xln, L * L, DP, 0, 0);
  gemm(xln, DP, w_tri_qkv, FP(31), nullptr, nullptr, qkv, L * L, 3 * DP, DP, 3 * DP, 0, 0, 0);
  attn(qkv, ctx, L, L, HT, DP);
  gemm(ctx, DP, w_tri_out, FP(33), pair_cur, pair_cur, nullptr, L * L, DP, DP, DP, 0, 0, 0);

  // ---- outputs: [msa | pair] ----
  hipMemcpyAsync(d_out, msa_cur, (size_t)S * L * DM * 4, hipMemcpyDeviceToDevice, stream);
  hipMemcpyAsync((float*)d_out + (size_t)S * L * DM, pair_cur,
                 (size_t)L * L * DP * 4, hipMemcpyDeviceToDevice, stream);
  #undef FP
}
